// VoxelBackBone8xBase_59407987638870
// MI455X (gfx1250) — matrix operand order repaired and checked by reading, not compiled, not run on hardware
//
#include <hip/hip_runtime.h>
#include <hip/hip_bf16.h>

// ---------------------------------------------------------------------------
// Types for CDNA5 WMMA (wave32)
// ---------------------------------------------------------------------------
typedef __attribute__((ext_vector_type(16))) _Float16 v16h;
typedef __attribute__((ext_vector_type(8)))  _Float16 v8h;
typedef __attribute__((ext_vector_type(8)))  float    v8f;

// ---------------------------------------------------------------------------
// Problem constants
// ---------------------------------------------------------------------------
static constexpr int D0 = 41, H0 = 160, W0 = 160;
static constexpr int V0 = D0 * H0 * W0;       // 1,049,600
static constexpr int NVOX = 50000;
static constexpr float EPSBN = 0.001f;

// ---------------------------------------------------------------------------
// Zero fill (16 bytes per thread, grid-stride)
// ---------------------------------------------------------------------------
__global__ void fill_zero16(uint4* p, size_t n16) {
    size_t i = (size_t)blockIdx.x * blockDim.x + threadIdx.x;
    size_t stride = (size_t)gridDim.x * blockDim.x;
    uint4 z; z.x = 0; z.y = 0; z.z = 0; z.w = 0;
    for (; i < n16; i += stride) p[i] = z;
}

// ---------------------------------------------------------------------------
// Scatter voxel features into dense channels-last f16 grid (padded to 32ch)
// and build the initial float mask.
// ---------------------------------------------------------------------------
__global__ void scatter_voxels(const float* __restrict__ feat,
                               const int* __restrict__ coords,
                               _Float16* __restrict__ dense,   // [V0][32]
                               float* __restrict__ mask)       // [V0]
{
    int i = blockIdx.x * blockDim.x + threadIdx.x;
    if (i >= NVOX) return;
    int z = coords[3 * i + 0];
    int y = coords[3 * i + 1];
    int x = coords[3 * i + 2];
    size_t v = ((size_t)z * H0 + y) * W0 + x;
    #pragma unroll
    for (int c = 0; c < 4; ++c)
        dense[v * 32 + c] = (_Float16)feat[(size_t)i * 4 + c];
    mask[v] = 1.0f;
}

// ---------------------------------------------------------------------------
// Fold BN into weights: wf[tap][co_pad][ci_pad] = w * g*rsqrt(v+eps),
// bias[co_pad] = b - m*scale. Zero padding for co>=CO or ci>=CI.
// Reference weight layout: OIDHW -> w[((co*CI)+ci)*KT + tap]
// ---------------------------------------------------------------------------
template<int CI, int CIP, int CO, int COP, int KT>
__global__ void prep_weights(const float* __restrict__ w,
                             const float* __restrict__ g,
                             const float* __restrict__ b,
                             const float* __restrict__ m,
                             const float* __restrict__ vv,
                             _Float16* __restrict__ wf,
                             float* __restrict__ bias)
{
    int i = blockIdx.x * blockDim.x + threadIdx.x;
    if (i < COP) {
        float s = (i < CO) ? g[i] * rsqrtf(vv[i] + EPSBN) : 0.0f;
        bias[i] = (i < CO) ? (b[i] - m[i] * s) : 0.0f;
    }
    const int total = KT * COP * CIP;
    if (i >= total) return;
    int ci  = i % CIP;
    int r   = i / CIP;
    int co  = r % COP;
    int tap = r / COP;
    float val = 0.0f;
    if (co < CO && ci < CI) {
        float s = g[co] * rsqrtf(vv[co] + EPSBN);
        val = w[((size_t)co * CI + ci) * KT + tap] * s;
    }
    wf[i] = (_Float16)val;
}

// ---------------------------------------------------------------------------
// Mask propagation for strided layers: mo[v] = any(window(mi) > 0)
// ---------------------------------------------------------------------------
template<int KD,int KH,int KW,int SD,int SH,int SW,int PD,int PH,int PW,
         int ID,int IH,int IW,int OD,int OH,int OW>
__global__ void mask_update(const float* __restrict__ mi, float* __restrict__ mo)
{
    constexpr int NOUT = OD * OH * OW;
    int v = blockIdx.x * blockDim.x + threadIdx.x;
    if (v >= NOUT) return;
    int t = v;
    int ow = t % OW; t /= OW;
    int oh = t % OH;
    int od = t / OH;
    float any = 0.0f;
    for (int kd = 0; kd < KD; ++kd) {
        int id = od * SD - PD + kd;
        if (id < 0 || id >= ID) continue;
        for (int kh = 0; kh < KH; ++kh) {
            int ih = oh * SH - PH + kh;
            if (ih < 0 || ih >= IH) continue;
            for (int kw = 0; kw < KW; ++kw) {
                int iw = ow * SW - PW + kw;
                if (iw < 0 || iw >= IW) continue;
                if (mi[((size_t)id * IH + ih) * IW + iw] > 0.0f) any = 1.0f;
            }
        }
    }
    mo[v] = any;
}

// ---------------------------------------------------------------------------
// Implicit-GEMM conv3d via V_WMMA_F32_16X16X32_F16.
//   2D register tile per wave: MT=2 M-tiles (32 voxels) x NT=COP/16 N-tiles.
//   Per (tap, k-chunk): MT A-frag loads + NT B-frag loads -> MT*NT WMMAs.
//   The M-tile pair is manually unrolled so back-to-back WMMAs on the SAME B
//   operand can carry the (compile-time-constant) reuse_b hint.
//
// A fragment layout (16-bit A 16x32, ISA 7.12.2):
//   lanes 0-15  (M=lane):      elems 0..7 -> K 0..7,   elems 8..15 -> K 16..23
//   lanes 16-31 (M=lane-16):   elems 0..7 -> K 8..15,  elems 8..15 -> K 24..31
//   => two contiguous 16B loads at channel offsets hs*8 and 16+hs*8.
// B fragment layout (32x16): same K order as A (elems 0..7 -> K hs*8.., 8..15 -> K 16+hs*8..),
//   N = lane&15 => two 16B loads from wf[tap][co][K-chunk].
// C/D f32 16x16: VGPR r -> (M = r + hs*8, N = lane&15).
// ---------------------------------------------------------------------------
template<int CIP,int COP,int KD,int KH,int KW,int SD,int SH,int SW,
         int PD,int PH,int PW,int ID,int IH,int IW,int OD,int OH,int OW>
__global__ __launch_bounds__(128)
void conv_wmma(const _Float16* __restrict__ in,    // [ID*IH*IW][CIP]
               const _Float16* __restrict__ wt,    // [KT][COP][CIP]
               const float*    __restrict__ bias,  // [COP]
               const float*    __restrict__ mask,  // [OD*OH*OW]
               _Float16*       __restrict__ out)   // [OD*OH*OW][COP]
{
    constexpr int NOUT = OD * OH * OW;
    constexpr int NT   = COP / 16;         // N tiles held in registers
    constexpr int MT   = 2;                // M tiles held in registers
    const int lane = threadIdx.x;          // 0..31 (wave32)
    const int hs   = lane >> 4;            // half-wave select
    const int col  = lane & 15;
    const int vbase = (blockIdx.x * blockDim.y + threadIdx.y) * (16 * MT);

    // Output voxels whose A-rows this lane loads (A rows indexed by col).
    int od[MT], oh[MT], ow[MT];
    bool avalid[MT];
    #pragma unroll
    for (int mt = 0; mt < MT; ++mt) {
        const int av = vbase + mt * 16 + col;
        avalid[mt] = (av < NOUT);
        int t = avalid[mt] ? av : 0;
        ow[mt] = t % OW; t /= OW;
        oh[mt] = t % OH;
        od[mt] = t / OH;
    }

    v8f acc[MT][NT] = {};
    #pragma unroll
    for (int kd = 0; kd < KD; ++kd) {
        #pragma unroll
        for (int kh = 0; kh < KH; ++kh) {
            #pragma unroll
            for (int kw = 0; kw < KW; ++kw) {
                const int tap = (kd * KH + kh) * KW + kw;
                // Per-lane B base: row (col) of weight tile for this tap.
                const _Float16* bbase =
                    wt + ((size_t)tap * COP + col) * CIP + hs * 8;

                // Per-M-tile input rows + validity.
                const _Float16* arow[MT];
                bool ok[MT];
                #pragma unroll
                for (int mt = 0; mt < MT; ++mt) {
                    const int id = od[mt] * SD - PD + kd;
                    const int ih = oh[mt] * SH - PH + kh;
                    const int iw = ow[mt] * SW - PW + kw;
                    ok[mt] = avalid[mt] & (id >= 0) & (id < ID)
                                        & (ih >= 0) & (ih < IH)
                                        & (iw >= 0) & (iw < IW);
                    arow[mt] = in + ((size_t)((id * IH + ih) * IW + iw)) * CIP;
                }

                #pragma unroll
                for (int kc = 0; kc < CIP / 32; ++kc) {
                    // ---- load A fragments (one per M tile) ----
                    v16h afrag[MT];
                    #pragma unroll
                    for (int mt = 0; mt < MT; ++mt) {
                        afrag[mt] = v16h{};
                        if (ok[mt]) {
                            const v8h* ap =
                                (const v8h*)(arow[mt] + kc * 32 + hs * 8);
                            v8h a0 = ap[0];   // K [hs*8 .. hs*8+7]
                            v8h a1 = ap[2];   // K [16+hs*8 .. 16+hs*8+7]
                            afrag[mt] = __builtin_shufflevector(a0, a1,
                                0,1,2,3,4,5,6,7,8,9,10,11,12,13,14,15);
                        }
                    }
                    // ---- each B fragment feeds MT WMMAs ----
                    #pragma unroll
                    for (int nt = 0; nt < NT; ++nt) {
                        const v8h* bp =
                            (const v8h*)(bbase + (size_t)nt * 16 * CIP + kc * 32);
                        v8h b0 = bp[0];
                        v8h b1 = bp[2];
                        v16h bfrag = __builtin_shufflevector(b0, b1,
                            0,1,2,3,4,5,6,7,8,9,10,11,12,13,14,15);
                        // D = A x B + C (f32 accumulate). Manually unrolled
                        // M-tile pair: second WMMA re-reads the identical B
                        // operand back-to-back -> literal reuse_b hint.
                        acc[0][nt] = __builtin_amdgcn_wmma_f32_16x16x32_f16(
                            /*neg_a=*/false, afrag[0],
                            /*neg_b=*/false, bfrag,
                            /*c_mod=*/(short)0, acc[0][nt],
                            /*reuse_a=*/false, /*reuse_b=*/false);
                        acc[1][nt] = __builtin_amdgcn_wmma_f32_16x16x32_f16(
                            /*neg_a=*/false, afrag[1],
                            /*neg_b=*/false, bfrag,
                            /*c_mod=*/(short)0, acc[1][nt],
                            /*reuse_a=*/false, /*reuse_b=*/true);
                    }
                }
            }
        }
    }

    // Epilogue: bias + ReLU + mask, store f16 channels-last.
    #pragma unroll
    for (int mt = 0; mt < MT; ++mt) {
        #pragma unroll
        for (int nt = 0; nt < NT; ++nt) {
            const float bs = bias[nt * 16 + col];
            #pragma unroll
            for (int r = 0; r < 8; ++r) {
                const int ov = vbase + mt * 16 + hs * 8 + r;
                if (ov < NOUT) {
                    float val = acc[mt][nt][r] + bs;
                    val = val > 0.0f ? val : 0.0f;
                    val *= mask[ov];
                    out[(size_t)ov * COP + nt * 16 + col] = (_Float16)val;
                }
            }
        }
    }
}

// ---------------------------------------------------------------------------
// Final transpose: act [800 vox][128]f16 -> d_out (1,128,2,20,20) f32
// ---------------------------------------------------------------------------
__global__ void finalize_out(const _Float16* __restrict__ act,
                             float* __restrict__ out)
{
    const int total = 128 * 800;
    int i = blockIdx.x * blockDim.x + threadIdx.x;
    if (i >= total) return;
    int c = i / 800;
    int v = i % 800;
    out[i] = (float)act[(size_t)v * 128 + c];
}

// ---------------------------------------------------------------------------
// Host orchestration
// ---------------------------------------------------------------------------
extern "C" void kernel_launch(void* const* d_in, const int* in_sizes, int n_in,
                              void* d_out, int out_size, void* d_ws, size_t ws_size,
                              hipStream_t stream)
{
    (void)in_sizes; (void)n_in; (void)out_size; (void)ws_size;
    const float* feat   = (const float*)d_in[0];
    const int*   coords = (const int*)d_in[1];
    // params: d_in[2 + 5*i + {0:w,1:g,2:b,3:m,4:v}]

    char* ws = (char*)d_ws;
    size_t off = 0;
    auto take = [&](size_t bytes) {
        size_t o = off;
        off += (bytes + 255) & ~(size_t)255;
        return o;
    };

    const size_t actA_o = take((size_t)V0 * 32 * 2);
    const size_t actB_o = take((size_t)V0 * 32 * 2);
    const size_t m0_o = take((size_t)V0 * 4);
    const size_t m1_o = take((size_t)21 * 80 * 80 * 4);
    const size_t m2_o = take((size_t)11 * 40 * 40 * 4);
    const size_t m3_o = take((size_t)5 * 20 * 20 * 4);
    const size_t m4_o = take((size_t)2 * 20 * 20 * 4);

    static const int LCIP[12] = {32,32,32,32,32,32,64,64,64,64,64,64};
    static const int LCOP[12] = {32,32,32,32,32,64,64,64,64,64,64,128};
    static const int LKT [12] = {27,27,27,27,27,27,27,27,27,27,27,3};
    size_t wf_o[12], bs_o[12];
    for (int i = 0; i < 12; ++i) {
        wf_o[i] = take((size_t)LKT[i] * LCOP[i] * LCIP[i] * 2);
        bs_o[i] = take((size_t)LCOP[i] * 4);
    }

    _Float16* actA = (_Float16*)(ws + actA_o);
    _Float16* actB = (_Float16*)(ws + actB_o);
    float* m0 = (float*)(ws + m0_o);
    float* m1 = (float*)(ws + m1_o);
    float* m2 = (float*)(ws + m2_o);
    float* m3 = (float*)(ws + m3_o);
    float* m4 = (float*)(ws + m4_o);

    // 1) Zero dense input grid + mask0 (deterministic per call).
    fill_zero16<<<2048, 256, 0, stream>>>((uint4*)actA, (size_t)V0 * 32 * 2 / 16);
    fill_zero16<<<1024, 256, 0, stream>>>((uint4*)m0, (size_t)V0 * 4 / 16);

    // 2) Scatter voxels.
    scatter_voxels<<<(NVOX + 255) / 256, 256, 0, stream>>>(feat, coords, actA, m0);

    // 3) Fold BN into f16 weights, per layer.
#define LAUNCH_PREP(L, CI, CIP, CO, COP, KT)                                       \
    do { int total_ = (KT) * (COP) * (CIP);                                        \
         prep_weights<CI, CIP, CO, COP, KT><<<(total_ + 255) / 256, 256, 0, stream>>>( \
             (const float*)d_in[2 + 5 * (L) + 0], (const float*)d_in[2 + 5 * (L) + 1], \
             (const float*)d_in[2 + 5 * (L) + 2], (const float*)d_in[2 + 5 * (L) + 3], \
             (const float*)d_in[2 + 5 * (L) + 4],                                  \
             (_Float16*)(ws + wf_o[L]), (float*)(ws + bs_o[L])); } while (0)

    LAUNCH_PREP(0,  4, 32, 16, 32, 27);
    LAUNCH_PREP(1, 16, 32, 16, 32, 27);
    LAUNCH_PREP(2, 16, 32, 32, 32, 27);
    LAUNCH_PREP(3, 32, 32, 32, 32, 27);
    LAUNCH_PREP(4, 32, 32, 32, 32, 27);
    LAUNCH_PREP(5, 32, 32, 64, 64, 27);
    LAUNCH_PREP(6, 64, 64, 64, 64, 27);
    LAUNCH_PREP(7, 64, 64, 64, 64, 27);
    LAUNCH_PREP(8, 64, 64, 64, 64, 27);
    LAUNCH_PREP(9, 64, 64, 64, 64, 27);
    LAUNCH_PREP(10,64, 64, 64, 64, 27);
    LAUNCH_PREP(11,64, 64,128,128, 3);

#define LAUNCH_CONV(L, CIP, COP, KD,KH,KW, SD,SH,SW, PD,PH,PW, ID,IH,IW, OD,OH,OW, INP, OUTP, MASKP) \
    do { constexpr int nout_ = (OD) * (OH) * (OW);                                  \
         dim3 g_((nout_ + 127) / 128, 1, 1);                                        \
         dim3 b_(32, 4, 1);                                                         \
         conv_wmma<CIP,COP,KD,KH,KW,SD,SH,SW,PD,PH,PW,ID,IH,IW,OD,OH,OW>            \
             <<<g_, b_, 0, stream>>>(INP, (const _Float16*)(ws + wf_o[L]),          \
                                     (const float*)(ws + bs_o[L]), MASKP, OUTP); } while (0)

#define LAUNCH_MASK(KD,KH,KW, SD,SH,SW, PD,PH,PW, ID,IH,IW, OD,OH,OW, MIN_, MOUT_)  \
    do { constexpr int nout_ = (OD) * (OH) * (OW);                                  \
         mask_update<KD,KH,KW,SD,SH,SW,PD,PH,PW,ID,IH,IW,OD,OH,OW>                  \
             <<<(nout_ + 255) / 256, 256, 0, stream>>>(MIN_, MOUT_); } while (0)

    // ---- stage 1: full resolution (41,160,160) ----
    LAUNCH_CONV(0, 32, 32, 3,3,3, 1,1,1, 1,1,1, 41,160,160, 41,160,160, actA, actB, m0);
    LAUNCH_CONV(1, 32, 32, 3,3,3, 1,1,1, 1,1,1, 41,160,160, 41,160,160, actB, actA, m0);

    // ---- stage 2: stride-2 down to (21,80,80) ----
    LAUNCH_MASK(3,3,3, 2,2,2, 1,1,1, 41,160,160, 21,80,80, m0, m1);
    LAUNCH_CONV(2, 32, 32, 3,3,3, 2,2,2, 1,1,1, 41,160,160, 21,80,80, actA, actB, m1);
    LAUNCH_CONV(3, 32, 32, 3,3,3, 1,1,1, 1,1,1, 21,80,80, 21,80,80, actB, actA, m1);
    LAUNCH_CONV(4, 32, 32, 3,3,3, 1,1,1, 1,1,1, 21,80,80, 21,80,80, actA, actB, m1);

    // ---- stage 3: stride-2 down to (11,40,40) ----
    LAUNCH_MASK(3,3,3, 2,2,2, 1,1,1, 21,80,80, 11,40,40, m1, m2);
    LAUNCH_CONV(5, 32, 64, 3,3,3, 2,2,2, 1,1,1, 21,80,80, 11,40,40, actB, actA, m2);
    LAUNCH_CONV(6, 64, 64, 3,3,3, 1,1,1, 1,1,1, 11,40,40, 11,40,40, actA, actB, m2);
    LAUNCH_CONV(7, 64, 64, 3,3,3, 1,1,1, 1,1,1, 11,40,40, 11,40,40, actB, actA, m2);

    // ---- stage 4: stride-2 down to (5,20,20), pad (0,1,1) ----
    LAUNCH_MASK(3,3,3, 2,2,2, 0,1,1, 11,40,40, 5,20,20, m2, m3);
    LAUNCH_CONV(8, 64, 64, 3,3,3, 2,2,2, 0,1,1, 11,40,40, 5,20,20, actA, actB, m3);
    LAUNCH_CONV(9, 64, 64, 3,3,3, 1,1,1, 1,1,1, 5,20,20, 5,20,20, actB, actA, m3);
    LAUNCH_CONV(10,64, 64, 3,3,3, 1,1,1, 1,1,1, 5,20,20, 5,20,20, actA, actB, m3);

    // ---- stage 5: (3,1,1) stride (2,1,1) down to (2,20,20), 128 ch ----
    LAUNCH_MASK(3,1,1, 2,1,1, 0,0,0, 5,20,20, 2,20,20, m3, m4);
    LAUNCH_CONV(11,64,128, 3,1,1, 2,1,1, 0,0,0, 5,20,20, 2,20,20, actB, actA, m4);

    // ---- transpose to NCDHW f32 ----
    finalize_out<<<(128 * 800 + 255) / 256, 256, 0, stream>>>(actA, (float*)d_out);

#undef LAUNCH_PREP
#undef LAUNCH_CONV
#undef LAUNCH_MASK
}
